// MultiHeadAttention_154618823122
// MI455X (gfx1250) — matrix operand order repaired and checked by reading, not compiled, not run on hardware
//
#include <hip/hip_runtime.h>

// ---------------------------------------------------------------------------
// Multi-head attention forward for gfx1250 (MI455X), bf16 WMMA + fp32 softmax.
// D_MODEL=1024, HEADS=16, DEPTH=64, B=2, S=2048.
// Pipeline: qkv_proj -> scores -> softmax -> ctx -> out_proj.
// ---------------------------------------------------------------------------

typedef __attribute__((ext_vector_type(16))) __bf16 v16bf;
typedef __attribute__((ext_vector_type(8)))  __bf16 v8bf;
typedef __attribute__((ext_vector_type(4)))  __bf16 v4bf;
typedef __attribute__((ext_vector_type(8)))  float  v8f;
typedef int v4i __attribute__((vector_size(4 * sizeof(int))));

#define DM    1024
#define NH    16
#define DH    64
#define BATCH 2
#define SEQ   2048

// GEMM tile config: 4 wave32 per block, 64x64 tile, K-step 32.
#define BM 64
#define BN 64
#define BK 32
#define NT 128
#define LDA (BK + 8)   // padded LDS row stride for A tiles: 80B (16B aligned)
#define LDB (BN + 8)   // padded LDS row stride for B tiles: 144B (16B aligned)

// ---- CDNA5 async global->LDS copy (guarded; falls back to b128 copy) ------
#if defined(__has_builtin)
#if __has_builtin(__builtin_amdgcn_global_load_async_to_lds_b128) && \
    __has_builtin(__builtin_amdgcn_s_wait_asynccnt)
#define USE_ASYNC_LDS 1
#endif
#endif
#ifndef USE_ASYNC_LDS
#define USE_ASYNC_LDS 0
#endif

__device__ __forceinline__ void copy16_g2l(const __bf16* g, __bf16* l) {
#if USE_ASYNC_LDS
  typedef __attribute__((address_space(1))) v4i Gv4;
  typedef __attribute__((address_space(3))) v4i Lv4;
  __builtin_amdgcn_global_load_async_to_lds_b128(
      (Gv4*)(uintptr_t)g, (Lv4*)(unsigned int)(uintptr_t)l, 0, 0);
#else
  *(v8bf*)l = *(const v8bf*)g;
#endif
}

__device__ __forceinline__ void async_join() {
#if USE_ASYNC_LDS
  __builtin_amdgcn_s_wait_asynccnt(0);
#endif
}

// fp32 -> bf16 packed convert of a float4
__device__ __forceinline__ v4bf cvt4(const float4 f) {
  v4bf o;
  o[0] = (__bf16)f.x; o[1] = (__bf16)f.y; o[2] = (__bf16)f.z; o[3] = (__bf16)f.w;
  return o;
}

// ---- WMMA fragment loaders (per documented gfx1250 lane layouts) ----------

// A (16x32 bf16): lane holds row m = lane%16; K halves {0..7,16..23} for
// lanes 0-15, {8..15,24..31} for lanes 16-31; elem e -> vgpr e/2, half e%2.
__device__ __forceinline__ v16bf load_a_frag(const __bf16* As, int wm, int mi, int lane) {
  const __bf16* row = As + (wm + mi * 16 + (lane & 15)) * LDA;
  const int kbase = (lane < 16) ? 0 : 8;
  v16bf a;
#pragma unroll
  for (int e = 0; e < 16; ++e) {
    int vg = e >> 1, hh = e & 1;
    int k = (vg < 4) ? (kbase + vg * 2 + hh) : (16 + kbase + (vg - 4) * 2 + hh);
    a[e] = row[k];
  }
  return a;
}

// B (32x16 bf16): lane holds column n = lane%16; K halves {0..7,16..23} for
// lanes 0-15, {8..15,24..31} for lanes 16-31 (same order as A).
__device__ __forceinline__ v16bf load_b_frag(const __bf16* Bs, int wn, int ni, int lane) {
  const __bf16* p = Bs + wn + ni * 16 + (lane & 15);
  const int kbase = (lane < 16) ? 0 : 8;
  v16bf b;
#pragma unroll
  for (int e = 0; e < 16; ++e) {
    int k = (e < 8) ? (kbase + e) : (16 + kbase + (e - 8));
    b[e] = p[k * LDB];
  }
  return b;
}

__device__ __forceinline__ void mma_step(const __bf16* As, const __bf16* Bs,
                                         int wm, int wn, int lane, v8f acc[2][2]) {
  v16bf a0 = load_a_frag(As, wm, 0, lane);
  v16bf a1 = load_a_frag(As, wm, 1, lane);
  v16bf b0 = load_b_frag(Bs, wn, 0, lane);
  v16bf b1 = load_b_frag(Bs, wn, 1, lane);
  acc[0][0] = __builtin_amdgcn_wmma_f32_16x16x32_bf16(false, a0, false, b0, (short)0, acc[0][0], false, false);
  acc[0][1] = __builtin_amdgcn_wmma_f32_16x16x32_bf16(false, a0, false, b1, (short)0, acc[0][1], false, false);
  acc[1][0] = __builtin_amdgcn_wmma_f32_16x16x32_bf16(false, a1, false, b0, (short)0, acc[1][0], false, false);
  acc[1][1] = __builtin_amdgcn_wmma_f32_16x16x32_bf16(false, a1, false, b1, (short)0, acc[1][1], false, false);
}

// ---- Stage 1: fused Q/K/V projection (z picks which), heads layout out ----
__global__ __launch_bounds__(NT) void qkv_proj_kernel(
    const float* __restrict__ q, const float* __restrict__ k, const float* __restrict__ v,
    const float* __restrict__ wq, const float* __restrict__ bq,
    const float* __restrict__ wk, const float* __restrict__ bk,
    const float* __restrict__ wv, const float* __restrict__ bv,
    __bf16* __restrict__ Qb, __bf16* __restrict__ Kb, __bf16* __restrict__ Vb)
{
  __shared__ __bf16 As[BM * LDA];
  __shared__ __bf16 Bs[BK * LDB];
  const int z = blockIdx.z;
  const float* X    = (z == 0) ? q  : (z == 1) ? k  : v;
  const float* W    = (z == 0) ? wq : (z == 1) ? wk : wv;
  const float* bias = (z == 0) ? bq : (z == 1) ? bk : bv;
  __bf16* dst       = (z == 0) ? Qb : (z == 1) ? Kb : Vb;

  const int m0 = blockIdx.x * BM, n0 = blockIdx.y * BN;
  const int tid = threadIdx.x, lane = tid & 31, wid = tid >> 5;
  const int wm = (wid & 1) * 32, wn = (wid >> 1) * 32;
  v8f acc[2][2] = {};

  for (int k0 = 0; k0 < DM; k0 += BK) {
#pragma unroll
    for (int i = 0; i < (BM * BK) / (NT * 4); ++i) {   // 4 float4/thread
      int l = tid + NT * i;
      int r = l >> 3, c = (l & 7) * 4;
      *(v4bf*)&As[r * LDA + c] = cvt4(*(const float4*)&X[(size_t)(m0 + r) * DM + k0 + c]);
    }
#pragma unroll
    for (int i = 0; i < (BK * BN) / (NT * 4); ++i) {
      int l = tid + NT * i;
      int r = l >> 4, c = (l & 15) * 4;
      *(v4bf*)&Bs[r * LDB + c] = cvt4(*(const float4*)&W[(size_t)(k0 + r) * DM + n0 + c]);
    }
    if (k0 + BK < DM) {  // global_prefetch_b8 of next K tile
      __builtin_prefetch(&X[(size_t)(m0 + (tid >> 1)) * DM + k0 + BK], 0, 0);
      __builtin_prefetch(&W[(size_t)(k0 + BK + (tid >> 2)) * DM + n0 + (tid & 3) * 16], 0, 0);
    }
    __syncthreads();
    mma_step(As, Bs, wm, wn, lane, acc);
    __syncthreads();
  }

#pragma unroll
  for (int mi = 0; mi < 2; ++mi)
#pragma unroll
    for (int ni = 0; ni < 2; ++ni) {
      int n = n0 + wn + ni * 16 + (lane & 15);
      int h = n >> 6, d = n & 63;
      float bi = bias[n];
#pragma unroll
      for (int r = 0; r < 8; ++r) {
        int m = m0 + wm + mi * 16 + r + ((lane >> 4) << 3);
        int bb = m >> 11, s = m & (SEQ - 1);
        dst[((size_t)(bb * NH + h) * SEQ + s) * DH + d] = (__bf16)(acc[mi][ni][r] + bi);
      }
    }
}

// ---- Stage 2: scores = scale * Qh @ Kh^T + mask*-1e9 (pre-softmax) --------
__global__ __launch_bounds__(NT) void scores_kernel(
    const __bf16* __restrict__ Qb, const __bf16* __restrict__ Kb,
    const float* __restrict__ mask, float* __restrict__ attn)
{
  __shared__ __bf16 As[BM * LDA];
  __shared__ __bf16 Bs[BK * LDB];
  const int bh = blockIdx.z, b = bh >> 4;
  const __bf16* Qh = Qb + (size_t)bh * SEQ * DH;
  const __bf16* Kh = Kb + (size_t)bh * SEQ * DH;
  float* out = attn + (size_t)bh * SEQ * SEQ;
  const int m0 = blockIdx.x * BM, n0 = blockIdx.y * BN;
  const int tid = threadIdx.x, lane = tid & 31, wid = tid >> 5;
  const int wm = (wid & 1) * 32, wn = (wid >> 1) * 32;
  v8f acc[2][2] = {};

  for (int k0 = 0; k0 < DH; k0 += BK) {   // 2 iterations (DH=64)
#pragma unroll
    for (int i = 0; i < (BM * BK) / (NT * 8); ++i) {  // async DMA, 2 chunks/thread
      int l = tid + NT * i;
      int r = l >> 2, c = (l & 3) * 8;
      copy16_g2l(&Qh[(size_t)(m0 + r) * DH + k0 + c], &As[r * LDA + c]);
    }
#pragma unroll
    for (int i = 0; i < (BK * BN) / NT; ++i) {  // transpose K on LDS stage
      int l = tid + NT * i;
      int kk = l >> 5, d = l & 31;
      Bs[d * LDB + kk] = Kh[(size_t)(n0 + kk) * DH + k0 + d];
    }
    async_join();
    __syncthreads();
    mma_step(As, Bs, wm, wn, lane, acc);
    __syncthreads();
  }

  const float scale = 0.125f;  // 1/sqrt(64)
#pragma unroll
  for (int mi = 0; mi < 2; ++mi)
#pragma unroll
    for (int ni = 0; ni < 2; ++ni) {
      int n = n0 + wn + ni * 16 + (lane & 15);
      float mv = mask[b * SEQ + n] * -1e9f;
#pragma unroll
      for (int r = 0; r < 8; ++r) {
        int m = m0 + wm + mi * 16 + r + ((lane >> 4) << 3);
        out[(size_t)m * SEQ + n] = acc[mi][ni][r] * scale + mv;
      }
    }
}

// ---- Stage 3: in-place rowwise softmax over last dim (2048) ---------------
__global__ __launch_bounds__(256) void softmax_kernel(float* __restrict__ attn)
{
  __shared__ float red[256];
  float* p = attn + (size_t)blockIdx.x * SEQ;
  const int tid = threadIdx.x;
  float vals[SEQ / 256];
  float mx = -3.4e38f;
#pragma unroll
  for (int j = 0; j < SEQ / 256; ++j) {
    vals[j] = p[tid + 256 * j];
    mx = fmaxf(mx, vals[j]);
  }
  red[tid] = mx; __syncthreads();
  for (int s = 128; s > 0; s >>= 1) {
    if (tid < s) red[tid] = fmaxf(red[tid], red[tid + s]);
    __syncthreads();
  }
  mx = red[0]; __syncthreads();

  float sum = 0.f;
#pragma unroll
  for (int j = 0; j < SEQ / 256; ++j) {
    vals[j] = __expf(vals[j] - mx);
    sum += vals[j];
  }
  red[tid] = sum; __syncthreads();
  for (int s = 128; s > 0; s >>= 1) {
    if (tid < s) red[tid] += red[tid + s];
    __syncthreads();
  }
  const float inv = 1.0f / red[0];
#pragma unroll
  for (int j = 0; j < SEQ / 256; ++j) p[tid + 256 * j] = vals[j] * inv;
}

// ---- Stage 4: ctx = attn @ Vh, merged-heads bf16 output -------------------
__global__ __launch_bounds__(NT) void ctx_kernel(
    const float* __restrict__ attn, const __bf16* __restrict__ Vb, __bf16* __restrict__ Cb)
{
  __shared__ __bf16 As[BM * LDA];
  __shared__ __bf16 Bs[BK * LDB];
  const int bh = blockIdx.z, b = bh >> 4, h = bh & 15;
  const float* Ah = attn + (size_t)bh * SEQ * SEQ;
  const __bf16* Vh = Vb + (size_t)bh * SEQ * DH;
  const int m0 = blockIdx.x * BM;  // n0 == 0, BN == DH
  const int tid = threadIdx.x, lane = tid & 31, wid = tid >> 5;
  const int wm = (wid & 1) * 32, wn = (wid >> 1) * 32;
  v8f acc[2][2] = {};

  for (int k0 = 0; k0 < SEQ; k0 += BK) {
#pragma unroll
    for (int i = 0; i < (BK * BN) / (NT * 8); ++i) {  // async DMA of V tile
      int l = tid + NT * i;
      int r = l >> 3, c = (l & 7) * 8;
      copy16_g2l(&Vh[(size_t)(k0 + r) * DH + c], &Bs[r * LDB + c]);
    }
#pragma unroll
    for (int i = 0; i < (BM * BK) / (NT * 4); ++i) {  // fp32 attn -> bf16
      int l = tid + NT * i;
      int r = l >> 3, c = (l & 7) * 4;
      *(v4bf*)&As[r * LDA + c] = cvt4(*(const float4*)&Ah[(size_t)(m0 + r) * SEQ + k0 + c]);
    }
    if (k0 + BK < SEQ) {
      __builtin_prefetch(&Ah[(size_t)(m0 + (tid >> 1)) * SEQ + k0 + BK], 0, 0);
      __builtin_prefetch(&Vh[(size_t)(k0 + BK + (tid >> 2)) * DH + (tid & 3) * 16], 0, 0);
    }
    async_join();
    __syncthreads();
    mma_step(As, Bs, wm, wn, lane, acc);
    __syncthreads();
  }

#pragma unroll
  for (int mi = 0; mi < 2; ++mi)
#pragma unroll
    for (int ni = 0; ni < 2; ++ni) {
      int n = wn + ni * 16 + (lane & 15);
#pragma unroll
      for (int r = 0; r < 8; ++r) {
        int m = m0 + wm + mi * 16 + r + ((lane >> 4) << 3);
        Cb[((size_t)b * SEQ + m) * DM + h * DH + n] = (__bf16)acc[mi][ni][r];
      }
    }
}

// ---- Stage 5: out = ctx @ wo + bo (fp32) ----------------------------------
__global__ __launch_bounds__(NT) void out_proj_kernel(
    const __bf16* __restrict__ Cb, const float* __restrict__ wo,
    const float* __restrict__ bo, float* __restrict__ out)
{
  __shared__ __bf16 As[BM * LDA];
  __shared__ __bf16 Bs[BK * LDB];
  const int m0 = blockIdx.x * BM, n0 = blockIdx.y * BN;
  const int tid = threadIdx.x, lane = tid & 31, wid = tid >> 5;
  const int wm = (wid & 1) * 32, wn = (wid >> 1) * 32;
  v8f acc[2][2] = {};

  for (int k0 = 0; k0 < DM; k0 += BK) {
#pragma unroll
    for (int i = 0; i < (BM * BK) / (NT * 8); ++i) {  // async DMA of ctx tile
      int l = tid + NT * i;
      int r = l >> 2, c = (l & 3) * 8;
      copy16_g2l(&Cb[(size_t)(m0 + r) * DM + k0 + c], &As[r * LDA + c]);
    }
#pragma unroll
    for (int i = 0; i < (BK * BN) / (NT * 4); ++i) {  // fp32 wo -> bf16
      int l = tid + NT * i;
      int r = l >> 4, c = (l & 15) * 4;
      *(v4bf*)&Bs[r * LDB + c] = cvt4(*(const float4*)&wo[(size_t)(k0 + r) * DM + n0 + c]);
    }
    if (k0 + BK < DM) {
      __builtin_prefetch(&Cb[(size_t)(m0 + (tid >> 1)) * DM + k0 + BK], 0, 0);
      __builtin_prefetch(&wo[(size_t)(k0 + BK + (tid >> 2)) * DM + n0 + (tid & 3) * 16], 0, 0);
    }
    async_join();
    __syncthreads();
    mma_step(As, Bs, wm, wn, lane, acc);
    __syncthreads();
  }

#pragma unroll
  for (int mi = 0; mi < 2; ++mi)
#pragma unroll
    for (int ni = 0; ni < 2; ++ni) {
      int n = n0 + wn + ni * 16 + (lane & 15);
      float bi = bo[n];
#pragma unroll
      for (int r = 0; r < 8; ++r) {
        int m = m0 + wm + mi * 16 + r + ((lane >> 4) << 3);
        out[(size_t)m * DM + n] = acc[mi][ni][r] + bi;
      }
    }
}

// ---------------------------------------------------------------------------
extern "C" void kernel_launch(void* const* d_in, const int* in_sizes, int n_in,
                              void* d_out, int out_size, void* d_ws, size_t ws_size,
                              hipStream_t stream) {
  (void)in_sizes; (void)n_in; (void)out_size; (void)ws_size;
  const float* q    = (const float*)d_in[0];
  const float* k    = (const float*)d_in[1];
  const float* v    = (const float*)d_in[2];
  const float* mask = (const float*)d_in[3];
  const float* wq   = (const float*)d_in[4];
  const float* bq   = (const float*)d_in[5];
  const float* wk   = (const float*)d_in[6];
  const float* bk   = (const float*)d_in[7];
  const float* wv   = (const float*)d_in[8];
  const float* bv   = (const float*)d_in[9];
  const float* wo   = (const float*)d_in[10];
  const float* bo   = (const float*)d_in[11];

  float* out_f = (float*)d_out;                                // [B,S,DM]
  float* attn  = out_f + (size_t)BATCH * SEQ * DM;             // [B,H,S,S]

  const size_t headElems = (size_t)BATCH * NH * SEQ * DH;      // 4.19M
  __bf16* Qb = (__bf16*)d_ws;
  __bf16* Kb = Qb + headElems;
  __bf16* Vb = Kb + headElems;
  __bf16* Cb = Vb + headElems;                                 // [B,S,DM] bf16

  const dim3 blk(NT);
  qkv_proj_kernel<<<dim3((BATCH * SEQ) / BM, DM / BN, 3), blk, 0, stream>>>(
      q, k, v, wq, bq, wk, bk, wv, bv, Qb, Kb, Vb);
  scores_kernel<<<dim3(SEQ / BM, SEQ / BN, BATCH * NH), blk, 0, stream>>>(
      Qb, Kb, mask, attn);
  softmax_kernel<<<dim3(BATCH * NH * SEQ), dim3(256), 0, stream>>>(attn);
  ctx_kernel<<<dim3(SEQ / BM, 1, BATCH * NH), blk, 0, stream>>>(attn, Vb, Cb);
  out_proj_kernel<<<dim3((BATCH * SEQ) / BM, DM / BN, 1), blk, 0, stream>>>(
      Cb, wo, bo, out_f);
}
